// MoMNet_6141803233542
// MI455X (gfx1250) — matrix operand order repaired and checked by reading, not compiled, not run on hardware
//
#include <hip/hip_runtime.h>

// ---------------- problem constants ----------------
#define B_ROWS 16384
#define D_INN  256
#define D_OUTT 256
#define NEXP   8
#define C_HID  512
#define E_HID  1024
#define TM     32                      // rows per workgroup tile
#define TILES_PER_E (B_ROWS / TM)      // 512

typedef _Float16 h8  __attribute__((ext_vector_type(8)));
typedef _Float16 h16 __attribute__((ext_vector_type(16)));
typedef float    f8  __attribute__((ext_vector_type(8)));

// A-fragment: lane<->M, halves = K groups {hig..hig+7, 16+hig..16+hig+7}
__device__ __forceinline__ h16 mkfragA(const _Float16* p)
{
    h8 lo = *(const h8*)(p);
    h8 hi = *(const h8*)(p + 16);
    h16 a;
#pragma unroll
    for (int i = 0; i < 8; ++i) { a[i] = lo[i]; a[i + 8] = hi[i]; }
    return a;
}

// ---------------- WMMA block GEMM ----------------
// Computes 32 rows (2 M-strips) x NT*16 cols: acc[2][NT].
// As : LDS f16 activations, row-major, stride sA halves (16B aligned, +8 pad)
// Wc : global f16 weights, row-major [Kdim][Ndim]; B-frag: lane<->N, elements<->K (same K groups as A)
template<int NT>
__device__ __forceinline__ void gemm_block(
    const _Float16* __restrict__ As, int sA,
    const _Float16* __restrict__ Wc, int Kdim, int Ndim, int nbase,
    f8 acc[2][NT])
{
    const int lane = threadIdx.x & 31;
    const int m    = lane & 15;
    const int hig  = (lane >> 4) * 8;
    const _Float16* arow0 = As + m * sA + hig;           // rows 0..15
    const _Float16* arow1 = As + (16 + m) * sA + hig;    // rows 16..31
    const _Float16* brow  = Wc + (size_t)(lane >> 4) * 8 * Ndim + nbase + (lane & 15);
    for (int k = 0; k < Kdim; k += 32) {
        h16 a0 = mkfragA(arow0 + k);
        h16 a1 = mkfragA(arow1 + k);
        const _Float16* bp = brow + (size_t)k * Ndim;
#pragma unroll
        for (int t = 0; t < NT; ++t) {
            h16 b;                                       // one B per 2 WMMA
#pragma unroll
            for (int i = 0; i < 8; ++i) {
                b[i]     = bp[(size_t)i * Ndim + t * 16];
                b[i + 8] = bp[(size_t)(16 + i) * Ndim + t * 16];
            }
            acc[0][t] = __builtin_amdgcn_wmma_f32_16x16x32_f16(
                false, a0, false, b, (short)0, acc[0][t], false, false);
            acc[1][t] = __builtin_amdgcn_wmma_f32_16x16x32_f16(
                false, a1, false, b, (short)0, acc[1][t], false, false);
        }
    }
}

template<int NT>
__device__ __forceinline__ void zero_acc(f8 acc[2][NT])
{
#pragma unroll
    for (int s = 0; s < 2; ++s)
#pragma unroll
        for (int t = 0; t < NT; ++t)
#pragma unroll
            for (int i = 0; i < 8; ++i) acc[s][t][i] = 0.0f;
}

// bias + optional LeakyReLU(0.2), f32 -> f16 row-major store to LDS
template<int NT>
__device__ __forceinline__ void store_block_lds(
    _Float16* __restrict__ Os, int sO, int nbase,
    const float* __restrict__ bias, const f8 acc[2][NT], bool lrelu)
{
    const int lane = threadIdx.x & 31;
    const int n    = lane & 15;
    const int mofs = (lane >> 4) * 8;
#pragma unroll
    for (int s = 0; s < 2; ++s)
#pragma unroll
        for (int tt = 0; tt < NT; ++tt) {
            float bb = bias[nbase + tt * 16 + n];
#pragma unroll
            for (int v = 0; v < 8; ++v) {
                float val = acc[s][tt][v] + bb;
                if (lrelu) val = fmaxf(val, 0.2f * val);
                Os[(s * 16 + mofs + v) * sO + nbase + tt * 16 + n] = (_Float16)val;
            }
        }
}

// ---------------- weight convert fp32 -> f16 (layout preserved) ----------------
__global__ void cvt_f32_f16(const float* __restrict__ in, _Float16* __restrict__ out, long n)
{
    long i = (long)blockIdx.x * blockDim.x + threadIdx.x;
    long stride = (long)gridDim.x * blockDim.x;
    for (; i < n; i += stride) out[i] = (_Float16)in[i];
}

// ---------------- classifier: logits + argmax ----------------
__global__ void __launch_bounds__(256)
classifier_kernel(const float* __restrict__ x,
                  const _Float16* __restrict__ cW1c,  // f16 [256][512]
                  const float* __restrict__ cb1,
                  const float* __restrict__ cW2,      // f32 [512][8]
                  const float* __restrict__ cb2,
                  int* __restrict__ labels)
{
    extern __shared__ _Float16 smem[];
    _Float16* xt = smem;                         // 32 x 264
    _Float16* Hs = smem + 32 * 264;              // 32 x 520
    float*    lg = (float*)(Hs + 32 * 520);      // 32 x 8

    const int tid = threadIdx.x;
    const int rowbase = blockIdx.x * TM;

    { // load + convert x tile
        int r  = tid >> 3;
        int c0 = (tid & 7) * 32;
        const float* xr = x + (size_t)(rowbase + r) * D_INN + c0;
#pragma unroll
        for (int i = 0; i < 32; i += 4) {
            float4 v = *(const float4*)(xr + i);
            xt[r * 264 + c0 + i + 0] = (_Float16)v.x;
            xt[r * 264 + c0 + i + 1] = (_Float16)v.y;
            xt[r * 264 + c0 + i + 2] = (_Float16)v.z;
            xt[r * 264 + c0 + i + 3] = (_Float16)v.w;
        }
    }
    __syncthreads();

    const int w = tid >> 5;                      // wave 0..7 owns cols w*64..+63
    {
        int nbase = w * 64;                      // 8 waves x 64 = 512 cols, 1 pass
        f8 acc[2][4];
        zero_acc<4>(acc);
        gemm_block<4>(xt, 264, cW1c, D_INN, C_HID, nbase, acc);
        store_block_lds<4>(Hs, 520, nbase, cb1, acc, true);
    }
    __syncthreads();

    { // logits: H[32][512] @ cW2[512][8]
        int r = tid >> 3, e = tid & 7;
        float s = cb2[e];
        const _Float16* hr = Hs + r * 520;
        for (int k = 0; k < C_HID; ++k) s += (float)hr[k] * cW2[k * NEXP + e];
        lg[r * NEXP + e] = s;
    }
    __syncthreads();
    if (tid < TM) {
        const float* l = lg + tid * NEXP;
        int bi = 0; float bv = l[0];
#pragma unroll
        for (int e = 1; e < NEXP; ++e) if (l[e] > bv) { bv = l[e]; bi = e; }
        labels[rowbase + tid] = bi;
    }
}

// ---------------- routing ----------------
__global__ void zero_counts_kernel(int* __restrict__ counts)
{
    if (threadIdx.x < NEXP) counts[threadIdx.x] = 0;
}

__global__ void count_kernel(const int* __restrict__ labels, int* __restrict__ counts, int n)
{
    int i = blockIdx.x * blockDim.x + threadIdx.x;
    if (i < n) atomicAdd(&counts[labels[i]], 1);
}

__global__ void scan_kernel(const int* __restrict__ counts,
                            int* __restrict__ bases, int* __restrict__ fill)
{
    if (threadIdx.x == 0 && blockIdx.x == 0) {
        int run = 0;
        for (int e = 0; e < NEXP; ++e) { bases[e] = run; fill[e] = run; run += counts[e]; }
    }
}

__global__ void scatter_kernel(const int* __restrict__ labels, int* __restrict__ fill,
                               int* __restrict__ rowidx, int n)
{
    int i = blockIdx.x * blockDim.x + threadIdx.x;
    if (i < n) {
        int p = atomicAdd(&fill[labels[i]], 1);
        rowidx[p] = i;
    }
}

// ---------------- routed 3-layer expert MLP ----------------
__global__ void __launch_bounds__(256)
expert_kernel(const float* __restrict__ x,
              const _Float16* __restrict__ W1c, const float* __restrict__ b1,
              const _Float16* __restrict__ W2c, const float* __restrict__ b2,
              const _Float16* __restrict__ W3c, const float* __restrict__ b3,
              const int* __restrict__ rowidx, const int* __restrict__ counts,
              const int* __restrict__ bases, float* __restrict__ out)
{
    const int e = blockIdx.x >> 9;               // / TILES_PER_E
    const int t = blockIdx.x & (TILES_PER_E - 1);
    const int cnt = counts[e];
    if (t * TM >= cnt) return;                   // block-uniform exit
    const int base = bases[e];

    extern __shared__ _Float16 smem[];
    _Float16* buf0 = smem;                       // 32 x 1032 (x tile uses stride 264)
    _Float16* buf1 = smem + 32 * 1032;           // 32 x 1032
    __shared__ int gr[TM];

    const int tid = threadIdx.x;
    if (tid < TM) {
        int idx = t * TM + tid;
        gr[tid] = rowidx[base + (idx < cnt ? idx : 0)];  // pad with a valid row
    }
    __syncthreads();

    { // gather + convert x rows
        int r  = tid >> 3;
        int c0 = (tid & 7) * 32;
        const float* xr = x + (size_t)gr[r] * D_INN + c0;
#pragma unroll
        for (int i = 0; i < 32; i += 4) {
            float4 v = *(const float4*)(xr + i);
            buf0[r * 264 + c0 + i + 0] = (_Float16)v.x;
            buf0[r * 264 + c0 + i + 1] = (_Float16)v.y;
            buf0[r * 264 + c0 + i + 2] = (_Float16)v.z;
            buf0[r * 264 + c0 + i + 3] = (_Float16)v.w;
        }
    }
    __syncthreads();

    const int w = tid >> 5;                      // wave owns a 64-col group per pass
    const _Float16* w1 = W1c + (size_t)e * D_INN * E_HID;
    const _Float16* w2 = W2c + (size_t)e * E_HID * E_HID;
    const _Float16* w3 = W3c + (size_t)e * E_HID * D_OUTT;
    const float* bb1 = b1 + e * E_HID;
    const float* bb2 = b2 + e * E_HID;
    const float* bb3 = b3 + e * D_OUTT;

    // layer 1: buf0 (x, K=256) -> buf1 (h1, N=1024): 2 passes of 512 cols
    for (int p = 0; p < 2; ++p) {
        int nbase = p * 512 + w * 64;
        f8 acc[2][4];
        zero_acc<4>(acc);
        gemm_block<4>(buf0, 264, w1, D_INN, E_HID, nbase, acc);
        store_block_lds<4>(buf1, 1032, nbase, bb1, acc, true);
    }
    __syncthreads();

    // layer 2: buf1 (h1, K=1024) -> buf0 (h2, N=1024)
    for (int p = 0; p < 2; ++p) {
        int nbase = p * 512 + w * 64;
        f8 acc[2][4];
        zero_acc<4>(acc);
        gemm_block<4>(buf1, 1032, w2, E_HID, E_HID, nbase, acc);
        store_block_lds<4>(buf0, 1032, nbase, bb2, acc, true);
    }
    __syncthreads();

    // layer 3: buf0 (h2, K=1024) -> out (N=256): 8 waves x 32 cols, scatter fp32
    {
        int nbase = w * 32;
        f8 acc[2][2];
        zero_acc<2>(acc);
        gemm_block<2>(buf0, 1032, w3, E_HID, D_OUTT, nbase, acc);

        const int lane = tid & 31;
        const int n    = lane & 15;
        const int mofs = (lane >> 4) * 8;
#pragma unroll
        for (int s = 0; s < 2; ++s)
#pragma unroll
            for (int tt = 0; tt < 2; ++tt) {
                int col = nbase + tt * 16 + n;
                float bbv = bb3[col];
#pragma unroll
                for (int v = 0; v < 8; ++v) {
                    int rl = s * 16 + mofs + v;
                    if (t * TM + rl < cnt)
                        out[(size_t)gr[rl] * D_OUTT + col] = acc[s][tt][v] + bbv;
                }
            }
    }
}

// ---------------- host ----------------
extern "C" void kernel_launch(void* const* d_in, const int* in_sizes, int n_in,
                              void* d_out, int out_size, void* d_ws, size_t ws_size,
                              hipStream_t stream)
{
    const float* x   = (const float*)d_in[0];
    const float* cW1 = (const float*)d_in[1];
    const float* cb1 = (const float*)d_in[2];
    const float* cW2 = (const float*)d_in[3];
    const float* cb2 = (const float*)d_in[4];
    const float* W1  = (const float*)d_in[5];
    const float* b1  = (const float*)d_in[6];
    const float* W2  = (const float*)d_in[7];
    const float* b2  = (const float*)d_in[8];
    const float* W3  = (const float*)d_in[9];
    const float* b3  = (const float*)d_in[10];
    float* out = (float*)d_out;

    char* ws = (char*)d_ws;
    _Float16* cW1c = (_Float16*)(ws + 0);          // 262,144 B
    _Float16* W1c  = (_Float16*)(ws + 262144);     // 4,194,304 B
    _Float16* W2c  = (_Float16*)(ws + 4456448);    // 16,777,216 B
    _Float16* W3c  = (_Float16*)(ws + 21233664);   // 4,194,304 B
    int* labels = (int*)(ws + 25427968);           // 65,536 B
    int* counts = (int*)(ws + 25493504);           // 32 B
    int* bases  = (int*)(ws + 25493536);           // 32 B
    int* fill   = (int*)(ws + 25493568);           // 32 B
    int* rowidx = (int*)(ws + 25493600);           // 65,536 B

    const int cls_lds = 32 * 264 * 2 + 32 * 520 * 2 + TM * NEXP * 4;   // 51,200 B
    const int exp_lds = 2 * 32 * 1032 * 2;                             // 132,096 B
    (void)hipFuncSetAttribute(reinterpret_cast<const void*>(classifier_kernel),
                              hipFuncAttributeMaxDynamicSharedMemorySize, cls_lds);
    (void)hipFuncSetAttribute(reinterpret_cast<const void*>(expert_kernel),
                              hipFuncAttributeMaxDynamicSharedMemorySize, exp_lds);

    // 1) weights fp32 -> f16 (layout preserved)
    cvt_f32_f16<<<512, 256, 0, stream>>>(cW1, cW1c, (long)D_INN * C_HID);
    cvt_f32_f16<<<1024, 256, 0, stream>>>(W1, W1c, (long)NEXP * D_INN * E_HID);
    cvt_f32_f16<<<2048, 256, 0, stream>>>(W2, W2c, (long)NEXP * E_HID * E_HID);
    cvt_f32_f16<<<1024, 256, 0, stream>>>(W3, W3c, (long)NEXP * E_HID * D_OUTT);

    // 2) classifier + argmax
    classifier_kernel<<<B_ROWS / TM, 256, cls_lds, stream>>>(x, cW1c, cb1, cW2, cb2, labels);

    // 3) routing
    zero_counts_kernel<<<1, 32, 0, stream>>>(counts);
    count_kernel<<<B_ROWS / 256, 256, 0, stream>>>(labels, counts, B_ROWS);
    scan_kernel<<<1, 1, 0, stream>>>(counts, bases, fill);
    scatter_kernel<<<B_ROWS / 256, 256, 0, stream>>>(labels, fill, rowidx, B_ROWS);

    // 4) routed experts (blocks beyond each expert's count exit immediately)
    expert_kernel<<<NEXP * TILES_PER_E, 256, exp_lds, stream>>>(
        x, W1c, b1, W2c, b2, W3c, b3, rowidx, counts, bases, out);
}
